// LIFLayer_41927470743712
// MI455X (gfx1250) — matrix operand order repaired and checked by reading, not compiled, not run on hardware
//
#include <hip/hip_runtime.h>
#include <hip/hip_bf16.h>

typedef __attribute__((ext_vector_type(16))) __bf16 v16bf;
typedef __attribute__((ext_vector_type(8)))  __bf16 v8bf;
typedef __attribute__((ext_vector_type(8)))  float  v8f;

#define T_STEPS 500
#define BATCH   64
#define C_IN    512
#define C_OUT   512
#define ALPHA_LIF 0.951229424500714f   // exp(-0.001/0.02)

// ---------------------------------------------------------------------------
// Kernel 1: temporal scan. U[t,b,k] = x_t + trace_t, trace_t = a*trace + x_t.
// One thread per (b,k); sequential over t (true recurrence, elementwise only).
// Output stored bf16 (WMMA A operand precision).
// ---------------------------------------------------------------------------
__global__ void lif_trace_scan(const float* __restrict__ x,       // [T,B,C_IN]
                               const float* __restrict__ alpha_p, // 1 elem
                               __bf16* __restrict__ U)            // [T,B,C_IN] bf16
{
    int idx = blockIdx.x * blockDim.x + threadIdx.x;   // b*C_IN + k
    if (idx >= BATCH * C_IN) return;
    float a = alpha_p[0];
    a = fminf(fmaxf(a, 0.0f), 0.9999f);
    float trace = 0.0f;
    for (int t = 0; t < T_STEPS; ++t) {
        float xt = x[(size_t)t * (BATCH * C_IN) + idx];
        trace = a * trace + xt;
        U[(size_t)t * (BATCH * C_IN) + idx] = (__bf16)(xt + trace);
    }
}

// ---------------------------------------------------------------------------
// Kernel 2: persistent LIF-GEMM. One wave per 16x16 output tile (4 B-tiles x
// 32 N-tiles = 128 waves). W column slice (512x16) preloaded once into 16
// register-resident B-fragments; membrane potential v lives in the C-layout
// accumulator registers across all 500 steps.
// __launch_bounds__(32,1) + waves_per_eu(1): single-wave block may use the
// full VGPR file (prevents the ~48-VGPR spill of B fragments to scratch seen
// at default occupancy targets).
// ---------------------------------------------------------------------------
__global__ void
__launch_bounds__(32, 1)
__attribute__((amdgpu_waves_per_eu(1)))
lif_wmma(const __bf16* __restrict__ U,     // [T,B,C_IN] bf16
         const float*  __restrict__ W,     // [C_IN,C_OUT] f32
         float* __restrict__ spikes,       // [T,B,C_OUT]
         float* __restrict__ rates)        // [B,C_OUT]
{
    const int tile = blockIdx.x;        // 0..127
    const int tb   = tile >> 5;         // batch tile 0..3
    const int tn   = tile & 31;         // cout  tile 0..31
    const int lane = threadIdx.x;       // 0..31 (wave32)
    const int half = lane >> 4;         // 0 | 1
    const int l15  = lane & 15;

    // --- one-time: load W column slice into B fragments (ISA 32x16 layout:
    //     lane<16 -> column N=l15, K = {kbase..kbase+7, kbase+16..kbase+23}; lane>=16 -> +8 K offset)
    v16bf bfrag[16];
    const int ncol = tn * 16 + l15;
#pragma unroll
    for (int kk = 0; kk < 16; ++kk) {
        const int kbase = kk * 32 + half * 8;
        v16bf bv;
#pragma unroll
        for (int j = 0; j < 16; ++j)
            bv[j] = (__bf16)W[(size_t)(kbase + j + ((j >> 3) << 3)) * C_OUT + ncol];
        bfrag[kk] = bv;
    }

    // A operand addressing (ISA 16x32 layout: lane<16 -> row M=l15,
    // K = {kbase..kbase+7, kbase+16..kbase+23}; lane>=16 -> +8 K offset)
    const int arow = tb * 16 + l15;      // global batch row for this lane
    const int asel = half * 8;

    v8f v    = {};   // membrane potential, C/D layout (VGPR r: M=r | r+8, N=l15)
    v8f racc = {};   // spike-rate accumulator

    for (int t = 0; t < T_STEPS; ++t) {
        const __bf16* Ut = U + (size_t)t * (BATCH * C_IN) + (size_t)arow * C_IN;
        v8f c = {};
#pragma unroll
        for (int kk = 0; kk < 16; ++kk) {
            const int kbase = kk * 32 + asel;
            v8bf lo = *(const v8bf*)(Ut + kbase);        // K = kbase..kbase+7
            v8bf hi = *(const v8bf*)(Ut + kbase + 16);   // K = kbase+16..+23
            v16bf a;
#pragma unroll
            for (int j = 0; j < 8; ++j) { a[j] = lo[j]; a[8 + j] = hi[j]; }
            c = __builtin_amdgcn_wmma_f32_16x16x32_bf16(
                    /*neg_a=*/false, a, /*neg_b=*/false, bfrag[kk],
                    /*c_mod=*/(short)0, c, /*reuse_a=*/false, /*reuse_b=*/false);
        }

        if (t + 1 < T_STEPS)   // global_prefetch_b8 of next step's A rows
            __builtin_prefetch((const void*)(U + (size_t)(t + 1) * (BATCH * C_IN)
                                               + (size_t)arow * C_IN), 0, 3);

        // LIF update + SuperSpike forward threshold + hard reset
        float* sp = spikes + (size_t)t * (BATCH * C_OUT);
#pragma unroll
        for (int r = 0; r < 8; ++r) {
            float vn = ALPHA_LIF * v[r] + (1.0f - ALPHA_LIF) * c[r];
            const bool fire = (vn > 1.0f);           // spike(v - V_TH)
            const float s = fire ? 1.0f : 0.0f;
            v[r] = fire ? 0.0f : vn;                 // hard reset to V_RESET
            racc[r] += s;
            const int m = r + half * 8;              // C-layout row within tile
            sp[(size_t)(tb * 16 + m) * C_OUT + tn * 16 + l15] = s;
        }
    }

#pragma unroll
    for (int r = 0; r < 8; ++r) {
        const int m = r + half * 8;
        rates[(size_t)(tb * 16 + m) * C_OUT + tn * 16 + l15] =
            racc[r] * (1.0f / (float)T_STEPS);
    }
}

// ---------------------------------------------------------------------------
extern "C" void kernel_launch(void* const* d_in, const int* in_sizes, int n_in,
                              void* d_out, int out_size, void* d_ws, size_t ws_size,
                              hipStream_t stream)
{
    const float* x_seq   = (const float*)d_in[0];   // [500,64,512] f32
    const float* W       = (const float*)d_in[1];   // [512,512]    f32
    const float* alpha_p = (const float*)d_in[2];   // scalar

    float* spikes = (float*)d_out;                               // [500,64,512]
    float* rates  = spikes + (size_t)T_STEPS * BATCH * C_OUT;    // [64,512]

    __bf16* U = (__bf16*)d_ws;   // needs 500*64*512*2 = 32,768,000 bytes

    // Pass 1: elementwise temporal scan -> U (bf16)
    {
        const int n = BATCH * C_IN;                 // 32768 threads
        lif_trace_scan<<<(n + 255) / 256, 256, 0, stream>>>(x_seq, alpha_p, U);
    }
    // Pass 2: persistent WMMA LIF (128 single-wave blocks, wave32)
    {
        lif_wmma<<<128, 32, 0, stream>>>(U, W, spikes, rates);
    }
}
